// AtomTransformer_38525856645640
// MI455X (gfx1250) — matrix operand order repaired and checked by reading, not compiled, not run on hardware
//
#include <hip/hip_runtime.h>
#include <hip/hip_bf16.h>
#include <math.h>

// ---------------------------------------------------------------------------
// AtomTransformer (AF3 Alg.23 style) for MI455X / gfx1250, wave32 + WMMA.
// B=1, N=2048, C=128, CP=16, H=4, dh=32, NB=3, NQ=32, NK=128.
// Attention is block-local: query block m (32 rows) sees keys [32m-48,32m+80).
// pair_rep is read only inside that window (~16MB of 256MB), once, for all 3
// blocks. All dense math via v_wmma_f32_16x16x32_f16; LDS staged in fragment-
// register order (each lane loads its v16h with 2x ds_load_b128). GEMM tiles
// are 32x128 per workgroup so the A panel is read from global exactly once.
// ---------------------------------------------------------------------------

#define NTOK   2048
#define CDIM   128
#define CPDIM  16
#define NHEAD  4
#define DHEAD  32
#define NBLK   3
#define WINK   128
#define QBLK   32

typedef __attribute__((ext_vector_type(16))) _Float16 v16h;
typedef __attribute__((ext_vector_type(8)))  float    v8f;

#define ACT_NONE 0
#define ACT_SIG  1
#define ACT_SILU 2

__device__ __forceinline__ float fast_sigmoid(float v) {
  return __builtin_amdgcn_rcpf(1.0f + __expf(-v));
}

// --------------------------- row LayerNorm (width 128) ---------------------
__global__ __launch_bounds__(128) void ln_rows_kernel(const float* __restrict__ in,
                                                      float* __restrict__ out) {
  const int row = blockIdx.x;
  const int tid = threadIdx.x;
  __shared__ float red[128];
  float x = in[row * CDIM + tid];
  red[tid] = x;
  __syncthreads();
  for (int s = 64; s > 0; s >>= 1) { if (tid < s) red[tid] += red[tid + s]; __syncthreads(); }
  const float mean = red[0] * (1.0f / CDIM);
  __syncthreads();
  float d = x - mean;
  red[tid] = d * d;
  __syncthreads();
  for (int s = 64; s > 0; s >>= 1) { if (tid < s) red[tid] += red[tid + s]; __syncthreads(); }
  out[row * CDIM + tid] = d * rsqrtf(red[0] * (1.0f / CDIM) + 1e-5f);
}

// ----- fused adaptive LN:  a = sig1*LN(x)+b1 ,  at = sig2*LN(x)+b2 ---------
__global__ __launch_bounds__(128) void fused_ada_kernel(
    const float* __restrict__ x,
    const float* __restrict__ sig1, const float* __restrict__ b1,
    const float* __restrict__ sig2, const float* __restrict__ b2,
    float* __restrict__ a, float* __restrict__ at) {
  const int row = blockIdx.x;
  const int tid = threadIdx.x;
  const int idx = row * CDIM + tid;
  __shared__ float red[128];
  float v = x[idx];
  red[tid] = v;
  __syncthreads();
  for (int s = 64; s > 0; s >>= 1) { if (tid < s) red[tid] += red[tid + s]; __syncthreads(); }
  const float mean = red[0] * (1.0f / CDIM);
  __syncthreads();
  float d = v - mean;
  red[tid] = d * d;
  __syncthreads();
  for (int s = 64; s > 0; s >>= 1) { if (tid < s) red[tid] += red[tid + s]; __syncthreads(); }
  const float l = d * rsqrtf(red[0] * (1.0f / CDIM) + 1e-5f);
  a[idx]  = sig1[idx] * l + b1[idx];
  at[idx] = sig2[idx] * l + b2[idx];
}

// ---- pair bias: LN over CP=16 of windowed pair_rep, dotted with wb for all
// ---- NB blocks x H heads.   pbias layout: [NB*H][NTOK][WINK]
__global__ __launch_bounds__(256) void pair_bias_kernel(
    const float* __restrict__ pair, const float* __restrict__ wb,
    float* __restrict__ pbias) {
  __shared__ float wbs[NBLK * CPDIM * NHEAD];  // 192 floats
  const int m   = blockIdx.x;
  const int tid = threadIdx.x;
  if (tid < NBLK * CPDIM * NHEAD) wbs[tid] = wb[tid];
  __syncthreads();
  const int k0 = QBLK * m - 48;
  for (int t = tid; t < QBLK * WINK; t += 256) {
    const int iq  = t >> 7;
    const int jj  = t & (WINK - 1);
    const int row = m * QBLK + iq;
    const int j   = k0 + jj;
    // prefetch next iteration's pair line (global_prefetch_b8)
    {
      const int t2 = t + 256;
      if (t2 < QBLK * WINK) {
        const int row2 = m * QBLK + (t2 >> 7);
        const int j2   = k0 + (t2 & (WINK - 1));
        if (j2 >= 0 && j2 < NTOK)
          __builtin_prefetch(pair + ((size_t)row2 * NTOK + j2) * CPDIM, 0, 1);
      }
    }
    if (j < 0 || j >= NTOK) {
      for (int bh = 0; bh < NBLK * NHEAD; ++bh)
        pbias[((size_t)bh * NTOK + row) * WINK + jj] = -1e10f;
      continue;
    }
    const float4* p4 = (const float4*)(pair + ((size_t)row * NTOK + j) * CPDIM);
    float v[CPDIM];
    {
      float4 a0 = p4[0], a1 = p4[1], a2 = p4[2], a3 = p4[3];
      v[0]=a0.x; v[1]=a0.y; v[2]=a0.z; v[3]=a0.w;
      v[4]=a1.x; v[5]=a1.y; v[6]=a1.z; v[7]=a1.w;
      v[8]=a2.x; v[9]=a2.y; v[10]=a2.z; v[11]=a2.w;
      v[12]=a3.x; v[13]=a3.y; v[14]=a3.z; v[15]=a3.w;
    }
    float s = 0.f, s2 = 0.f;
#pragma unroll
    for (int q = 0; q < CPDIM; ++q) { s += v[q]; s2 += v[q] * v[q]; }
    const float mean = s * (1.0f / CPDIM);
    const float rs   = rsqrtf(s2 * (1.0f / CPDIM) - mean * mean + 1e-5f);
#pragma unroll
    for (int q = 0; q < CPDIM; ++q) v[q] = (v[q] - mean) * rs;
    for (int b = 0; b < NBLK; ++b)
      for (int h = 0; h < NHEAD; ++h) {
        float acc = 0.f;
#pragma unroll
        for (int q = 0; q < CPDIM; ++q) acc += v[q] * wbs[(b * CPDIM + q) * NHEAD + h];
        pbias[(((size_t)(b * NHEAD + h)) * NTOK + row) * WINK + jj] = acc;
      }
  }
}

// --------------------------- generic WMMA GEMM -----------------------------
// out[M,Nc] = emul? ⊙ act(A[M,K] @ W[K,Nc] + bias?) + eadd?
// Workgroup tile 32(M) x 128(N): A panel read once; 4 waves, each wave keeps
// 4 accumulators and issues 4 WMMAs per K-step from one shared A fragment.
//   A frag (16-bit 16x32): lane half hv, element e -> K = e + 16*hv
//   B frag (16-bit 32x16): lane half hv, element e -> K = e + 16*hv
template<int ACT, bool HAS_BIAS, bool HAS_EMUL, bool HAS_EADD>
__global__ __launch_bounds__(128) void gemm_wmma_kernel(
    const float* __restrict__ A, const float* __restrict__ W,
    const float* __restrict__ bias, const float* __restrict__ emul,
    const float* __restrict__ eadd, float* __restrict__ out,
    int M, int K, int Nc) {
  __shared__ __align__(32) _Float16 Af[2][16][2][16];  // [mt][m][hv][e]
  __shared__ __align__(32) _Float16 Bf[8][16][2][16];  // [ntile][n][hv][e]
  const int tid  = threadIdx.x;
  const int bx   = blockIdx.x;   // 128-col block
  const int by   = blockIdx.y;   // 32-row block
  const int wave = tid >> 5;
  const int lane = tid & 31;
  const int hv   = lane >> 4;
  const int lm   = lane & 15;
  const int mt   = wave >> 1;
  const int nq0  = (wave & 1) * 4;  // first of this wave's 4 n-tiles
  v8f accs[4] = {v8f{}, v8f{}, v8f{}, v8f{}};
  for (int k0 = 0; k0 < K; k0 += 32) {
    // ---- A tile 32x32 (256 float4, 2 per thread) -> A-frag layout ----
#pragma unroll
    for (int f = tid; f < 256; f += 128) {
      const int r  = f >> 3;
      const int c4 = (f & 7) * 4;
      const float4 av = *(const float4*)&A[(size_t)(by * 32 + r) * K + k0 + c4];
      const int g = c4 >> 3;
      _Float16* dst = &Af[r >> 4][r & 15][c4 >> 4][c4 & 15];
      dst[0] = (_Float16)av.x; dst[1] = (_Float16)av.y;
      dst[2] = (_Float16)av.z; dst[3] = (_Float16)av.w;
    }
    // ---- W tile 32x128 (1024 float4, 8 per thread) -> B-frag layout ----
#pragma unroll
    for (int f = tid; f < 1024; f += 128) {
      const int r  = f >> 5;          // K row 0..31
      const int c4 = (f & 31) * 4;    // col 0..124
      const float4 wv4 = *(const float4*)&W[(size_t)(k0 + r) * Nc + bx * 128 + c4];
      const int hvB = r >> 4, eB = r & 15;
      const int ntB = c4 >> 4, nB = c4 & 15;
      Bf[ntB][nB + 0][hvB][eB] = (_Float16)wv4.x;
      Bf[ntB][nB + 1][hvB][eB] = (_Float16)wv4.y;
      Bf[ntB][nB + 2][hvB][eB] = (_Float16)wv4.z;
      Bf[ntB][nB + 3][hvB][eB] = (_Float16)wv4.w;
    }
    __syncthreads();
    const v16h af = *(const v16h*)&Af[mt][lm][hv][0];
#pragma unroll
    for (int qd = 0; qd < 4; ++qd) {
      const v16h bf = *(const v16h*)&Bf[nq0 + qd][lm][hv][0];
      accs[qd] = __builtin_amdgcn_wmma_f32_16x16x32_f16(false, af, false, bf,
                                                        (short)0, accs[qd], false, false);
    }
    __syncthreads();
  }
  const int row0 = by * 32 + mt * 16;
#pragma unroll
  for (int qd = 0; qd < 4; ++qd) {
    const int col = bx * 128 + (nq0 + qd) * 16 + lm;
    float bv = 0.0f;
    if (HAS_BIAS) bv = bias[col];
#pragma unroll
    for (int r = 0; r < 8; ++r) {
      const int row = row0 + r + 8 * hv;
      float v = accs[qd][r] + bv;
      if (ACT == ACT_SIG)       v = fast_sigmoid(v);
      else if (ACT == ACT_SILU) v = v * fast_sigmoid(v);
      const size_t oi = (size_t)row * Nc + col;
      if (HAS_EMUL) v *= emul[oi];
      if (HAS_EADD) v += eadd[oi];
      out[oi] = v;
    }
  }
}

// -------- windowed attention: one (query-block m, head h) per workgroup ----
// logits = (q/sqrt(dh)) @ k^T + pbias ; softmax over 128-key window ;
// og = g ⊙ (probs @ v).  All LDS in fragment-register order.
__global__ __launch_bounds__(128) void attn_kernel(
    const float* __restrict__ qb, const float* __restrict__ kb,
    const float* __restrict__ vb, const float* __restrict__ gb,
    const float* __restrict__ pbias, float* __restrict__ og) {
  __shared__ __align__(32) _Float16 Qf[2][16][2][16];     // A frag: [mt][m][hv][e]
  __shared__ __align__(32) _Float16 Kf[8][16][2][16];     // B frag: [nt][key%16][hv][e]
  __shared__ __align__(32) _Float16 Vf[2][16][4][2][16];  // B frag: [nt][c%16][ks][hv][e]
  __shared__ __align__(32) _Float16 Pf[2][16][4][2][16];  // A frag: [mt][m][ks][hv][e]
  __shared__ float probs[QBLK][WINK + 4];
  const int m    = blockIdx.x;
  const int h    = blockIdx.y;
  const int tid  = threadIdx.x;
  const int wave = tid >> 5, lane = tid & 31, hv = lane >> 4, lm = lane & 15;
  const int qs0  = m * QBLK;
  const int k0   = QBLK * m - 48;
  const float scale = 0.17677669529663687f;  // 1/sqrt(32)

  // q tile -> A-frag layout (32x32 f32 = 256 float4)
#pragma unroll
  for (int f = tid; f < 256; f += 128) {
    const int r  = f >> 3;
    const int c4 = (f & 7) * 4;
    const float4 av = *(const float4*)&qb[(size_t)(qs0 + r) * CDIM + h * DHEAD + c4];
    const int g = c4 >> 3;
    _Float16* dst = &Qf[r >> 4][r & 15][c4 >> 4][c4 & 15];
    dst[0] = (_Float16)(av.x * scale); dst[1] = (_Float16)(av.y * scale);
    dst[2] = (_Float16)(av.z * scale); dst[3] = (_Float16)(av.w * scale);
  }
  // k/v window (128x32 f32 = 1024 float4 each)
#pragma unroll
  for (int f = tid; f < 1024; f += 128) {
    const int key = f >> 3;
    const int c4  = (f & 7) * 4;
    const int j   = k0 + key;
    float4 kv = {0.f, 0.f, 0.f, 0.f}, vv = {0.f, 0.f, 0.f, 0.f};
    if (j >= 0 && j < NTOK) {
      kv = *(const float4*)&kb[(size_t)j * CDIM + h * DHEAD + c4];
      vv = *(const float4*)&vb[(size_t)j * CDIM + h * DHEAD + c4];
    }
    {  // K: B-frag for logits (n = key, Kdim = channel)
      _Float16* dst = &Kf[key >> 4][key & 15][c4 >> 4][c4 & 15];
      dst[0] = (_Float16)kv.x; dst[1] = (_Float16)kv.y;
      dst[2] = (_Float16)kv.z; dst[3] = (_Float16)kv.w;
    }
    {  // V: B-frag for o-step (n = channel, Kdim = key)
      const int ks = key >> 5, hvB = (key >> 4) & 1, eB = key & 15;
      const int ntB = c4 >> 4, nB = c4 & 15;
      Vf[ntB][nB + 0][ks][hvB][eB] = (_Float16)vv.x;
      Vf[ntB][nB + 1][ks][hvB][eB] = (_Float16)vv.y;
      Vf[ntB][nB + 2][ks][hvB][eB] = (_Float16)vv.z;
      Vf[ntB][nB + 3][ks][hvB][eB] = (_Float16)vv.w;
    }
  }
  __syncthreads();

  // logits: 2x8 tiles of 16x16, 4 per wave; pbias preloaded into accumulator.
#pragma unroll
  for (int t4 = 0; t4 < 4; ++t4) {
    const int tt = wave * 4 + t4;
    const int mt = tt >> 3, nt = tt & 7;
    const int col = nt * 16 + lm;
    v8f acc;
#pragma unroll
    for (int r = 0; r < 8; ++r)
      acc[r] = pbias[((size_t)h * NTOK + qs0 + mt * 16 + r + 8 * hv) * WINK + col];
    const v16h af = *(const v16h*)&Qf[mt][lm][hv][0];
    const v16h bf = *(const v16h*)&Kf[nt][lm][hv][0];
    acc = __builtin_amdgcn_wmma_f32_16x16x32_f16(false, af, false, bf,
                                                 (short)0, acc, false, false);
#pragma unroll
    for (int r = 0; r < 8; ++r)
      probs[mt * 16 + r + 8 * hv][col] = acc[r];
  }
  __syncthreads();

  // softmax over 128-key window (one thread per query row), results written
  // straight into the A-fragment layout for the probs@V WMMA.
  if (tid < QBLK) {
    float mx = -3.0e38f;
    for (int kk = 0; kk < WINK; ++kk) mx = fmaxf(mx, probs[tid][kk]);
    float sum = 0.f;
    for (int kk = 0; kk < WINK; ++kk) {
      const float ev = __expf(probs[tid][kk] - mx);
      probs[tid][kk] = ev;
      sum += ev;
    }
    const float inv = __builtin_amdgcn_rcpf(sum);
    const int m_ = tid & 15, mt_ = tid >> 4;
#pragma unroll
    for (int ks = 0; ks < 4; ++ks)
#pragma unroll
      for (int g2 = 0; g2 < 4; ++g2) {
        _Float16* dst = &Pf[mt_][m_][ks][g2 >> 1][(g2 & 1) * 8];
        const float* src = &probs[tid][ks * 32 + g2 * 8];
#pragma unroll
        for (int i = 0; i < 8; ++i) dst[i] = (_Float16)(src[i] * inv);
      }
  }
  __syncthreads();

  // o = probs @ v : 2x2 tiles, one per wave, K=128 in 4 steps of 32.
  {
    const int mt = wave >> 1, nt = wave & 1;
    v8f acc = {};
#pragma unroll
    for (int ks = 0; ks < 4; ++ks) {
      const v16h af = *(const v16h*)&Pf[mt][lm][ks][hv][0];
      const v16h bf = *(const v16h*)&Vf[nt][lm][ks][hv][0];
      acc = __builtin_amdgcn_wmma_f32_16x16x32_f16(false, af, false, bf,
                                                   (short)0, acc, false, false);
    }
    const int col = nt * 16 + lm;
#pragma unroll
    for (int r = 0; r < 8; ++r) {
      const int row = qs0 + mt * 16 + r + 8 * hv;
      const size_t gi = (size_t)row * CDIM + h * DHEAD + col;
      og[gi] = gb[gi] * acc[r];
    }
  }
}

// ---------------------------------------------------------------------------
template<int ACT, bool HAS_BIAS, bool HAS_EMUL, bool HAS_EADD>
static inline void launch_gemm(const float* A, const float* W, const float* bias,
                               const float* emul, const float* eadd, float* out,
                               int M, int K, int Nc, hipStream_t s) {
  dim3 grid(Nc / 128, M / 32);
  gemm_wmma_kernel<ACT, HAS_BIAS, HAS_EMUL, HAS_EADD>
      <<<grid, 128, 0, s>>>(A, W, bias, emul, eadd, out, M, K, Nc);
}

extern "C" void kernel_launch(void* const* d_in, const int* in_sizes, int n_in,
                              void* d_out, int out_size, void* d_ws, size_t ws_size,
                              hipStream_t stream) {
  (void)in_sizes; (void)n_in; (void)out_size; (void)ws_size;
  const float* q        = (const float*)d_in[0];
  const float* c        = (const float*)d_in[1];
  const float* pair     = (const float*)d_in[2];
  const float* ada1_gw  = (const float*)d_in[3];
  const float* ada1_gb  = (const float*)d_in[4];
  const float* ada1_bw  = (const float*)d_in[5];
  const float* wq       = (const float*)d_in[6];
  const float* bq       = (const float*)d_in[7];
  const float* wk       = (const float*)d_in[8];
  const float* wv       = (const float*)d_in[9];
  const float* wb       = (const float*)d_in[10];
  const float* wg       = (const float*)d_in[11];
  const float* wo       = (const float*)d_in[12];
  const float* sgw      = (const float*)d_in[13];
  const float* sgb      = (const float*)d_in[14];
  const float* ada2_gw  = (const float*)d_in[15];
  const float* ada2_gb  = (const float*)d_in[16];
  const float* ada2_bw  = (const float*)d_in[17];
  const float* w1       = (const float*)d_in[18];
  const float* w2       = (const float*)d_in[19];
  const float* w3       = (const float*)d_in[20];
  const float* tgw      = (const float*)d_in[21];
  const float* tgb      = (const float*)d_in[22];

  float* xout_final = (float*)d_out;

  // ---- workspace carve-up (floats) ----
  const size_t NC = (size_t)NTOK * CDIM;  // 262144
  float* ws = (float*)d_ws;
  size_t off = 0;
  float* sn       = ws + off; off += NC;
  float* sig1     = ws + off; off += NC;
  float* b1c      = ws + off; off += NC;
  float* sig2     = ws + off; off += NC;
  float* b2c      = ws + off; off += NC;
  float* sgate    = ws + off; off += NC;
  float* tgate    = ws + off; off += NC;
  float* abuf     = ws + off; off += NC;
  float* atbuf    = ws + off; off += NC;
  float* qbuf     = ws + off; off += NC;
  float* kbuf     = ws + off; off += NC;
  float* vbuf     = ws + off; off += NC;
  float* gbuf     = ws + off; off += NC;
  float* ogbuf    = ws + off; off += NC;
  float* attn_out = ws + off; off += NC;
  float* x0       = ws + off; off += NC;
  float* t1       = ws + off; off += (size_t)NTOK * 2 * CDIM;
  float* hidden   = ws + off; off += (size_t)NTOK * 2 * CDIM;
  float* pbias    = ws + off; off += (size_t)NBLK * NHEAD * NTOK * WINK;

  // ---- precompute: sn = LN(c); pair window bias for all blocks/heads ----
  ln_rows_kernel<<<NTOK, 128, 0, stream>>>(c, sn);
  pair_bias_kernel<<<NTOK / QBLK, 256, 0, stream>>>(pair, wb, pbias);

  // x starts as q (ping-pong so block 2 output lands in d_out)
  hipMemcpyAsync(x0, q, NC * sizeof(float), hipMemcpyDeviceToDevice, stream);

  for (int b = 0; b < NBLK; ++b) {
    const float* xin  = (b == 1) ? xout_final : x0;
    float*       xout = (b == 1) ? x0 : xout_final;
    const size_t wo128 = (size_t)b * CDIM * CDIM;
    const size_t wo256 = (size_t)b * CDIM * 2 * CDIM;
    const size_t bo    = (size_t)b * CDIM;

    // sn-conditioned gates/biases for this block
    launch_gemm<ACT_SIG, true, false, false>(sn, ada1_gw + wo128, ada1_gb + bo,
        nullptr, nullptr, sig1, NTOK, CDIM, CDIM, stream);
    launch_gemm<ACT_NONE, false, false, false>(sn, ada1_bw + wo128, nullptr,
        nullptr, nullptr, b1c, NTOK, CDIM, CDIM, stream);
    launch_gemm<ACT_SIG, true, false, false>(sn, ada2_gw + wo128, ada2_gb + bo,
        nullptr, nullptr, sig2, NTOK, CDIM, CDIM, stream);
    launch_gemm<ACT_NONE, false, false, false>(sn, ada2_bw + wo128, nullptr,
        nullptr, nullptr, b2c, NTOK, CDIM, CDIM, stream);
    launch_gemm<ACT_SIG, true, false, false>(sn, sgw + wo128, sgb + bo,
        nullptr, nullptr, sgate, NTOK, CDIM, CDIM, stream);
    launch_gemm<ACT_SIG, true, false, false>(sn, tgw + wo128, tgb + bo,
        nullptr, nullptr, tgate, NTOK, CDIM, CDIM, stream);

    // adaptive LN of x -> a (attention stream), at (transition stream)
    fused_ada_kernel<<<NTOK, 128, 0, stream>>>(xin, sig1, b1c, sig2, b2c, abuf, atbuf);

    // q/k/v/gate projections
    launch_gemm<ACT_NONE, true, false, false>(abuf, wq + wo128, bq + bo,
        nullptr, nullptr, qbuf, NTOK, CDIM, CDIM, stream);
    launch_gemm<ACT_NONE, false, false, false>(abuf, wk + wo128, nullptr,
        nullptr, nullptr, kbuf, NTOK, CDIM, CDIM, stream);
    launch_gemm<ACT_NONE, false, false, false>(abuf, wv + wo128, nullptr,
        nullptr, nullptr, vbuf, NTOK, CDIM, CDIM, stream);
    launch_gemm<ACT_SIG, false, false, false>(abuf, wg + wo128, nullptr,
        nullptr, nullptr, gbuf, NTOK, CDIM, CDIM, stream);

    // windowed attention with pair bias, gated output
    {
      dim3 grid(NTOK / QBLK, NHEAD);
      attn_kernel<<<grid, 128, 0, stream>>>(
          qbuf, kbuf, vbuf, gbuf,
          pbias + (size_t)b * NHEAD * NTOK * WINK, ogbuf);
    }

    // attn_out = sigmoid(sn@sgw+sgb) ⊙ (og @ wo)
    launch_gemm<ACT_NONE, false, true, false>(ogbuf, wo + wo128, nullptr,
        sgate, nullptr, attn_out, NTOK, CDIM, CDIM, stream);

    // conditioned SwiGLU transition + residual:
    // t1 = silu(at@w1); hidden = t1 ⊙ (at@w2); x = tg ⊙ (hidden@w3) + attn_out
    launch_gemm<ACT_SILU, false, false, false>(atbuf, w1 + wo256, nullptr,
        nullptr, nullptr, t1, NTOK, CDIM, 2 * CDIM, stream);
    launch_gemm<ACT_NONE, false, true, false>(atbuf, w2 + wo256, nullptr,
        t1, nullptr, hidden, NTOK, CDIM, 2 * CDIM, stream);
    launch_gemm<ACT_NONE, false, true, true>(hidden, w3 + wo256, nullptr,
        tgate, attn_out, xout, NTOK, 2 * CDIM, CDIM, stream);
  }
}
